// VectorQuantizer_12232066859402
// MI455X (gfx1250) — matrix operand order repaired and checked by reading, not compiled, not run on hardware
//
#include <hip/hip_runtime.h>
#include <hip/hip_bf16.h>
#include <math.h>

typedef __attribute__((ext_vector_type(16))) __bf16 v16bf;
typedef __attribute__((ext_vector_type(8)))  float  v8f;

#define DIMD   256
#define NUMK   4096
#define NBATCH 32
#define HWP    1024
#define NTOK   32768
#define NELEM  8388608

union FragU { uint4 u[2]; v16bf v; };

__device__ __forceinline__ unsigned short f2bf(float f) {
    unsigned u = __float_as_uint(f);
    unsigned r = u + 0x7FFFu + ((u >> 16) & 1u);   // round-to-nearest-even
    return (unsigned short)(r >> 16);
}
__device__ __forceinline__ float bf2f(unsigned short h) {
    return __uint_as_float(((unsigned)h) << 16);
}

// ---------------- Kernel 1: weight -> bf16 hi/lo + |e_k|^2 ; zero histogram ----
__global__ void k_prep_w(const float* __restrict__ w,
                         unsigned short* __restrict__ whi,
                         unsigned short* __restrict__ wlo,
                         float* __restrict__ wsum2,
                         unsigned* __restrict__ hist) {
    __shared__ float red[256];
    int k = blockIdx.x, t = threadIdx.x;
    float v = w[(size_t)k * DIMD + t];
    unsigned short h = f2bf(v);
    whi[(size_t)k * DIMD + t] = h;
    wlo[(size_t)k * DIMD + t] = f2bf(v - bf2f(h));
    red[t] = v * v;
    __syncthreads();
    for (int s = 128; s > 0; s >>= 1) { if (t < s) red[t] += red[t + s]; __syncthreads(); }
    if (t == 0) wsum2[k] = red[0];
    if (blockIdx.x < 16) hist[blockIdx.x * 256 + t] = 0u;   // fresh each launch
}

// ---------------- Kernel 2: x [B,D,HW] -> token-major bf16 hi/lo [N,D] ---------
__global__ void k_prep_x(const float* __restrict__ x,
                         unsigned short* __restrict__ xhi,
                         unsigned short* __restrict__ xlo) {
    __shared__ float tile[32][33];
    int tx = threadIdx.x, ty = threadIdx.y;
    int p0 = blockIdx.x * 32, d0 = blockIdx.y * 32, b = blockIdx.z;
#pragma unroll
    for (int i = 0; i < 4; i++) {
        int d = d0 + ty + i * 8;
        tile[ty + i * 8][tx] = x[((size_t)b * DIMD + d) * HWP + p0 + tx];
    }
    __syncthreads();
#pragma unroll
    for (int i = 0; i < 4; i++) {
        int p = p0 + ty + i * 8;
        int d = d0 + tx;
        float v = tile[tx][ty + i * 8];
        unsigned short h = f2bf(v);
        size_t off = (size_t)(b * HWP + p) * DIMD + d;
        xhi[off] = h;
        xlo[off] = f2bf(v - bf2f(h));
    }
}

// ---------------- Kernel 3: WMMA distance + argmin ----------------------------
// One workgroup = 8 waves = 128 tokens. Each wave: 16 tokens, A frags resident
// in VGPRs, loop over 256 chunks of 16 codes, 24 bf16 WMMAs per chunk.
// B fragments are explicitly double-buffered (incl. cross-chunk prefetch) so
// global loads overlap the WMMA chains instead of s_wait_loadcnt 0 stalls.
__global__ void __launch_bounds__(256)
k_argmin(const unsigned short* __restrict__ xhi, const unsigned short* __restrict__ xlo,
         const unsigned short* __restrict__ whi, const unsigned short* __restrict__ wlo,
         const float* __restrict__ wsum2,
         int* __restrict__ idxv, unsigned* __restrict__ hist,
         float* __restrict__ outIdx) {
    __shared__ int s_mini[128];
    const int tid  = threadIdx.x;
    const int lane = tid & 31, wv = tid >> 5;
    const int half = lane >> 4, l16 = lane & 15;

    // ---- load A fragments (16 tokens x 256 dims, hi+lo) into registers ----
    const int m = blockIdx.x * 128 + wv * 16 + l16;
    const unsigned short* xr_h = xhi + (size_t)m * DIMD;
    const unsigned short* xr_l = xlo + (size_t)m * DIMD;
    FragU ah[8], al[8];
#pragma unroll
    for (int s = 0; s < 8; s++) {
        const int d0 = s * 32;
        // A k order used here (matches B): lane holds contiguous k = half*16..+15
        ah[s].u[0] = *(const uint4*)(xr_h + d0 + half * 16);
        ah[s].u[1] = *(const uint4*)(xr_h + d0 + 8 + half * 16);
        al[s].u[0] = *(const uint4*)(xr_l + d0 + half * 16);
        al[s].u[1] = *(const uint4*)(xr_l + d0 + 8 + half * 16);
    }

    float vmin[8]; int imin[8];
#pragma unroll
    for (int r = 0; r < 8; r++) { vmin[r] = 3.0e38f; imin[r] = 0; }

    // B k order used here (matches A): lane holds contiguous k = half*16..+15 along D.
    // Row stride per 16-code chunk = 16*DIMD bf16 elements.
    const unsigned short* wr_h = whi + (size_t)l16 * DIMD + half * 16;
    const unsigned short* wr_l = wlo + (size_t)l16 * DIMD + half * 16;

    FragU bh[2], bl[2];
    // prefetch chunk 0, step 0
    bh[0].u[0] = *(const uint4*)(wr_h + 0);
    bh[0].u[1] = *(const uint4*)(wr_h + 8);
    bl[0].u[0] = *(const uint4*)(wr_l + 0);
    bl[0].u[1] = *(const uint4*)(wr_l + 8);

    for (int c = 0; c < NUMK / 16; c++) {
        const int n = c * 16 + l16;                 // this lane's code column
        const float wk = wsum2[n];
        v8f a0 = {}, a1 = {}, a2 = {};
#pragma unroll
        for (int s = 0; s < 8; s++) {
            const int cur = s & 1, nxt = cur ^ 1;
            // Prefetch next step (or next chunk's step 0; tail prefetch past the
            // last row lands in adjacent workspace — harmless, values unused).
            const int pre = (s < 7) ? (s + 1) * 32 : 16 * DIMD;
            bh[nxt].u[0] = *(const uint4*)(wr_h + pre);
            bh[nxt].u[1] = *(const uint4*)(wr_h + pre + 8);
            bl[nxt].u[0] = *(const uint4*)(wr_l + pre);
            bl[nxt].u[1] = *(const uint4*)(wr_l + pre + 8);
            // split-bf16 fp32 emulation: hi*hi + lo*hi + hi*lo
            a0 = __builtin_amdgcn_wmma_f32_16x16x32_bf16(false, ah[s].v, false, bh[cur].v,
                                                         (short)0, a0, false, false);
            a1 = __builtin_amdgcn_wmma_f32_16x16x32_bf16(false, al[s].v, false, bh[cur].v,
                                                         (short)0, a1, false, false);
            a2 = __builtin_amdgcn_wmma_f32_16x16x32_bf16(false, ah[s].v, false, bl[cur].v,
                                                         (short)0, a2, false, false);
        }
        wr_h += 16 * DIMD;
        wr_l += 16 * DIMD;
#pragma unroll
        for (int r = 0; r < 8; r++) {
            float sc = wk - 2.0f * (a0[r] + a1[r] + a2[r]);  // |e|^2 - 2 x.e
            if (sc < vmin[r]) { vmin[r] = sc; imin[r] = n; }
        }
    }

    // C layout: lane holds column n=l16; VGPR r is row m = r + half*8.
    // Reduce argmin across the 16 lanes of each half (masks < 16 stay in-half).
#pragma unroll
    for (int r = 0; r < 8; r++) {
        float v = vmin[r]; int ix = imin[r];
#pragma unroll
        for (int off = 1; off < 16; off <<= 1) {
            float ov = __shfl_xor(v, off);
            int   oi = __shfl_xor(ix, off);
            if (ov < v || (ov == v && oi < ix)) { v = ov; ix = oi; }
        }
        if (l16 == 0) s_mini[wv * 16 + r + half * 8] = ix;
    }
    __syncthreads();
    if (tid < 128) {
        int nTok = blockIdx.x * 128 + tid;
        int ix = s_mini[tid];
        idxv[nTok]   = ix;
        outIdx[nTok] = (float)ix;
        atomicAdd(&hist[ix], 1u);
    }
}

// ---------------- Kernel 4: gather + straight-through output + SSE partials ---
__global__ void k_gather(const float* __restrict__ x, const float* __restrict__ w,
                         const int* __restrict__ idxv, float* __restrict__ outq,
                         float* __restrict__ partials) {
    __shared__ float red[256];
    int t = threadIdx.x;
    int blk = blockIdx.x;
    int pblk = blk & 3;
    int d    = (blk >> 2) & 255;
    int b    = blk >> 10;
    int p = pblk * 256 + t;
    int n = b * HWP + p;
    int k = idxv[n];
    float q  = w[(size_t)k * DIMD + d];
    size_t off = ((size_t)b * DIMD + d) * HWP + p;
    float xv = x[off];
    float diff = q - xv;
    outq[off] = xv + diff;            // x + stop_grad(q - x)
    red[t] = diff * diff;
    __syncthreads();
    for (int s = 128; s > 0; s >>= 1) { if (t < s) red[t] += red[t + s]; __syncthreads(); }
    if (t == 0) partials[blk] = red[0];
}

// ---------------- Kernel 5: loss + perplexity (deterministic reductions) ------
__global__ void k_final(const float* __restrict__ partials,
                        const unsigned* __restrict__ hist,
                        float* __restrict__ outScalars) {
    __shared__ float red[256];
    int t = threadIdx.x;
    float s = 0.f;
    for (int i = t; i < 32768; i += 256) s += partials[i];
    red[t] = s; __syncthreads();
    for (int k = 128; k > 0; k >>= 1) { if (t < k) red[t] += red[t + k]; __syncthreads(); }
    float loss = 1.25f * red[0] / (float)NELEM;   // q_latent + 0.25 * e_latent
    float h = 0.f;
    for (int i = t; i < NUMK; i += 256) {
        float p = (float)hist[i] / (float)NTOK;
        h += p * logf(p + 1e-10f);
    }
    __syncthreads();
    red[t] = h; __syncthreads();
    for (int k = 128; k > 0; k >>= 1) { if (t < k) red[t] += red[t + k]; __syncthreads(); }
    if (t == 0) { outScalars[0] = loss; outScalars[1] = expf(-red[0]); }
}

// ---------------- Host launcher ----------------------------------------------
extern "C" void kernel_launch(void* const* d_in, const int* in_sizes, int n_in,
                              void* d_out, int out_size, void* d_ws, size_t ws_size,
                              hipStream_t stream) {
    const float* x = (const float*)d_in[0];   // [32,256,32,32]
    const float* w = (const float*)d_in[1];   // [4096,256]
    float* out = (float*)d_out;
    char* ws = (char*)d_ws;

    size_t o = 0;
    unsigned short* whi   = (unsigned short*)(ws + o); o += (size_t)NUMK * DIMD * 2;
    unsigned short* wlo   = (unsigned short*)(ws + o); o += (size_t)NUMK * DIMD * 2;
    float*          wsum2 = (float*)(ws + o);          o += (size_t)NUMK * 4;
    unsigned short* xhi   = (unsigned short*)(ws + o); o += (size_t)NTOK * DIMD * 2;
    unsigned short* xlo   = (unsigned short*)(ws + o); o += (size_t)NTOK * DIMD * 2;
    int*            idxv  = (int*)(ws + o);            o += (size_t)NTOK * 4;
    unsigned*       hist  = (unsigned*)(ws + o);       o += (size_t)NUMK * 4;
    float*          parts = (float*)(ws + o);          o += (size_t)32768 * 4;

    float* outQ       = out;                 // 8388608 floats
    float* outScalars = out + NELEM;         // loss, perplexity
    float* outIdx     = out + NELEM + 2;     // 32768 indices as float

    k_prep_w <<<dim3(NUMK),       dim3(256),     0, stream>>>(w, whi, wlo, wsum2, hist);
    k_prep_x <<<dim3(32, 8, 32),  dim3(32, 8),   0, stream>>>(x, xhi, xlo);
    k_argmin <<<dim3(NTOK / 128), dim3(256),     0, stream>>>(xhi, xlo, whi, wlo, wsum2,
                                                              idxv, hist, outIdx);
    k_gather <<<dim3(32768),      dim3(256),     0, stream>>>(x, w, idxv, outQ, parts);
    k_final  <<<dim3(1),          dim3(256),     0, stream>>>(parts, hist, outScalars);
}
